// MPNNEmbedding_45423574122974
// MI455X (gfx1250) — matrix operand order repaired and checked by reading, not compiled, not run on hardware
//
#include <hip/hip_runtime.h>

// ---------------------------------------------------------------------------
// MPNN embedding for MI455X (gfx1250, wave32, WMMA).
//
// Roofline: message GEMM = 3 x 65.5 GFLOP over ~5 GB of gather/scatter
// traffic (h kept bf16 for gathered operands). bf16 A/B + f32-accumulate
// v_wmma_f32_16x16x32_bf16 keeps us at the 23.3 TB/s memory roofline;
// f32 WMMA (16x16x4) has 8x less K/instr and would be compute bound.
// Aggregation via global_atomic_add_f32; GRU elementwise in f32.
// msg_W is streamed into LDS with double-buffered async-to-LDS loads
// (ASYNCcnt / s_wait_asynccnt) so B-chunk fetch overlaps WMMA issue.
// ---------------------------------------------------------------------------

#define NNODES 50000
#define NEDGES 800000
#define FNODE  64
#define FEDGE  16
#define HD     128      // node hidden
#define EHD    64       // edge hidden
#define KMSG   (2*HD + EHD)   // 320
#define NLAYER 3

typedef __bf16 bf16_t;
typedef bf16_t v16bf  __attribute__((ext_vector_type(16)));
typedef bf16_t bf16x2 __attribute__((ext_vector_type(2)));
typedef float  v8f    __attribute__((ext_vector_type(8)));

struct U8 { unsigned x[8]; };   // raw bits of one v16bf fragment (32 B)

// Native f32->bf16 converts (clang lowers to gfx1250 bf16 cvt ops).
__device__ __forceinline__ bf16_t f2bf(float f) { return (bf16_t)f; }
__device__ __forceinline__ unsigned pack2(float a, float b) {
  bf16x2 v; v[0] = (bf16_t)a; v[1] = (bf16_t)b;
  return __builtin_bit_cast(unsigned, v);
}
__device__ __forceinline__ v8f wmma_bf16(v16bf a, v16bf b, v8f c) {
  return __builtin_amdgcn_wmma_f32_16x16x32_bf16(false, a, false, b,
                                                 (short)0, c, false, false);
}
// A-matrix 16x32 bf16 lane layout (ISA 7.12.2): lane m=lane&15 holds row m;
// VGPR i (i<4): K = 2i + 8*hsel ; VGPR i (i>=4): K = 2(i-4)+16 + 8*hsel.
// => per lane + K-step, A frag = 16B at k0=32ks+8*hsel and 16B at k0+16.
__device__ __forceinline__ U8 a_frag_from2(uint4 c0, uint4 c1) {
  U8 u;
  u.x[0] = c0.x; u.x[1] = c0.y; u.x[2] = c0.z; u.x[3] = c0.w;
  u.x[4] = c1.x; u.x[5] = c1.y; u.x[6] = c1.z; u.x[7] = c1.w;
  return u;
}
// B-matrix 32x16 bf16: lane n=lane&15 holds col n; same K order as A
// => 16 B at K=8*hsel and 16 B at K=8*hsel+16 in [col][K] storage.
__device__ __forceinline__ int b_k(int hsel) { return hsel * 8; }

// LDS byte address of a __shared__ object: flat LDS-aperture addresses carry
// the LDS offset in addr[31:0] (ISA §10.2 aperture mapping).
__device__ __forceinline__ unsigned lds_addr32(const void* p) {
  return (unsigned)(unsigned long long)p;
}
// Async global->LDS 16-byte copy, tracked by ASYNCcnt (ISA §15.18.3 op 98).
__device__ __forceinline__ void async_copy_b128(unsigned lds_byte, const void* g) {
  asm volatile("global_load_async_to_lds_b128 %0, %1, off"
               :: "v"(lds_byte), "v"((unsigned long long)g) : "memory");
}
__device__ __forceinline__ void wait_async0() {
  asm volatile("s_wait_asynccnt 0" ::: "memory");
}

// ---------------------------------------------------------------------------
// Weight prep: f32 [K][Ncols] row-major  ->  bf16 [Ncols][Kpad] (zero padded)
// ---------------------------------------------------------------------------
__global__ void cvt_transpose_kernel(const float* __restrict__ src,
                                     bf16_t* __restrict__ dst,
                                     int K, int Ncols, int Kpad) {
  int idx = blockIdx.x * blockDim.x + threadIdx.x;
  if (idx >= Ncols * Kpad) return;
  int col = idx / Kpad, k = idx % Kpad;
  float v = (k < K) ? src[(long long)k * Ncols + col] : 0.0f;
  dst[idx] = f2bf(v);
}

// ---------------------------------------------------------------------------
// h = x @ node_W  ->  h_f32 [N,128], h_bf [N,128]; also zero agg.
// ---------------------------------------------------------------------------
__global__ __launch_bounds__(256)
void embed_nodes_kernel(const float* __restrict__ x,
                        const bf16_t* __restrict__ Wt,   // [128][64]
                        float* __restrict__ h,
                        bf16_t* __restrict__ h_bf,
                        float* __restrict__ agg) {
  __shared__ __align__(16) unsigned char sW[128 * 64 * 2];
  const int tid = threadIdx.x;
  {
    const uint4* s = (const uint4*)Wt; uint4* d = (uint4*)sW;
    for (int i = tid; i < 1024; i += 256) d[i] = s[i];
  }
  __syncthreads();
  const int wave = tid >> 5, lane = tid & 31, m = lane & 15, hsel = lane >> 4;
  const int tile = blockIdx.x * 8 + wave;
  if (tile >= NNODES / 16) return;
  const long long row = (long long)tile * 16 + m;
  const float* xr = x + row * FNODE;

  v8f acc[8];
#pragma unroll
  for (int nt = 0; nt < 8; ++nt) acc[nt] = v8f{};
#pragma unroll
  for (int ks = 0; ks < 2; ++ks) {
    const int k0 = ks * 32 + hsel * 8;
    float4 f0 = *(const float4*)(xr + k0);
    float4 f1 = *(const float4*)(xr + k0 + 4);
    float4 f2 = *(const float4*)(xr + k0 + 16);
    float4 f3 = *(const float4*)(xr + k0 + 20);
    U8 au;
    au.x[0] = pack2(f0.x, f0.y); au.x[1] = pack2(f0.z, f0.w);
    au.x[2] = pack2(f1.x, f1.y); au.x[3] = pack2(f1.z, f1.w);
    au.x[4] = pack2(f2.x, f2.y); au.x[5] = pack2(f2.z, f2.w);
    au.x[6] = pack2(f3.x, f3.y); au.x[7] = pack2(f3.z, f3.w);
    v16bf a = __builtin_bit_cast(v16bf, au);
#pragma unroll
    for (int nt = 0; nt < 8; ++nt) {
      const unsigned char* bp = sW + ((nt * 16 + m) * 64 + ks * 32 + b_k(hsel)) * 2;
      U8 bu = a_frag_from2(*(const uint4*)bp, *(const uint4*)(bp + 32));
      acc[nt] = wmma_bf16(a, __builtin_bit_cast(v16bf, bu), acc[nt]);
    }
  }
#pragma unroll
  for (int nt = 0; nt < 8; ++nt) {
    int col = nt * 16 + m;
#pragma unroll
    for (int v = 0; v < 8; ++v) {
      long long r = (long long)tile * 16 + v + 8 * hsel;
      float val = acc[nt][v];
      h[r * HD + col] = val;
      h_bf[r * HD + col] = f2bf(val);
      agg[r * HD + col] = 0.0f;
    }
  }
}

// ---------------------------------------------------------------------------
// e = edge_attr @ edge_W  ->  e_bf [E,64].  K=16 padded to 32.
// ---------------------------------------------------------------------------
__global__ __launch_bounds__(256)
void embed_edges_kernel(const float* __restrict__ ea,   // [E][16]
                        const bf16_t* __restrict__ Wt,  // [64][32] padded
                        bf16_t* __restrict__ e_bf) {
  __shared__ __align__(16) unsigned char sW[64 * 32 * 2];
  const int tid = threadIdx.x;
  {
    const uint4* s = (const uint4*)Wt; uint4* d = (uint4*)sW;
    for (int i = tid; i < 256; i += 256) d[i] = s[i];
  }
  __syncthreads();
  const int wave = tid >> 5, lane = tid & 31, m = lane & 15, hsel = lane >> 4;
  const long long tileBase = (long long)(blockIdx.x * 8 + wave) * 16;
  const float* ar = ea + (tileBase + m) * FEDGE;

  // A frag: K chunk1 (k0..k0+7) is real data, chunk2 (k0+16..) is pad = 0.
  const int k0 = hsel * 8;
  float4 f0 = *(const float4*)(ar + k0);
  float4 f1 = *(const float4*)(ar + k0 + 4);
  U8 au;
  au.x[0] = pack2(f0.x, f0.y); au.x[1] = pack2(f0.z, f0.w);
  au.x[2] = pack2(f1.x, f1.y); au.x[3] = pack2(f1.z, f1.w);
  au.x[4] = 0u; au.x[5] = 0u; au.x[6] = 0u; au.x[7] = 0u;
  v16bf a = __builtin_bit_cast(v16bf, au);

  v8f acc[4];
#pragma unroll
  for (int nt = 0; nt < 4; ++nt) {
    const unsigned char* bp = sW + ((nt * 16 + m) * 32 + b_k(hsel)) * 2;
    U8 bu = a_frag_from2(*(const uint4*)bp, *(const uint4*)(bp + 32));
    v8f z = v8f{};
    acc[nt] = wmma_bf16(a, __builtin_bit_cast(v16bf, bu), z);
  }
#pragma unroll
  for (int nt = 0; nt < 4; ++nt) {
    int col = nt * 16 + m;
#pragma unroll
    for (int v = 0; v < 8; ++v) {
      long long r = tileBase + v + 8 * hsel;
      e_bf[r * EHD + col] = f2bf(acc[nt][v]);
    }
  }
}

// ---------------------------------------------------------------------------
// Message GEMM + bias + scatter-add:
//   m[e] = concat(h_bf[src], e_bf[e], h_bf[dst]) @ msg_W + msg_b
//   agg[dst] += m[e]   (global_atomic_add_f32)
// One wave per 16-edge tile. B streamed via double-buffered async-to-LDS.
// ---------------------------------------------------------------------------
__global__ __launch_bounds__(256)
void msg_kernel(const bf16_t* __restrict__ h_bf,   // [N,128]
                const bf16_t* __restrict__ e_bf,   // [E,64]
                const bf16_t* __restrict__ Wt,     // [128][320]
                const float* __restrict__ bias,    // [128]
                const long long* __restrict__ ei,  // [2][E] int64
                float* __restrict__ agg) {
  __shared__ __align__(16) unsigned char sW[2 * 8192];  // 2 x 8 KB B chunks
  const int tid = threadIdx.x, wave = tid >> 5, lane = tid & 31;
  const int m = lane & 15, hsel = lane >> 4;
  const long long tileBase = (long long)(blockIdx.x * 8 + wave) * 16;
  const long long erow = tileBase + m;
  const long long s = ei[erow];
  const long long d = ei[NEDGES + erow];
  const bf16_t* hs = h_bf + s * HD;
  const bf16_t* hd = h_bf + d * HD;
  const bf16_t* ee = e_bf + erow * EHD;

  const unsigned sWlds = lds_addr32(sW);
  const int colT = tid >> 1, part = tid & 1;
  // Issue async load of B chunk ks into buffer b: [col][32 K] bf16 = 8 KB.
  auto issue_chunk = [&](int ks, int b) {
    const char* src = (const char*)Wt + colT * (KMSG * 2) + ks * 64 + part * 32;
    unsigned dst = sWlds + (unsigned)(b * 8192 + colT * 64 + part * 32);
    async_copy_b128(dst, src);
    async_copy_b128(dst + 16u, src + 16);
  };

  v8f acc[8];
#pragma unroll
  for (int nt = 0; nt < 8; ++nt) acc[nt] = v8f{};

  issue_chunk(0, 0);
  wait_async0();
  __syncthreads();

  for (int ks = 0; ks < KMSG / 32; ++ks) {
    const int cur = ks & 1;
    if (ks + 1 < KMSG / 32) issue_chunk(ks + 1, cur ^ 1);

    // A frag: two contiguous 16 B chunks of the concatenated row
    // (k0 = 32ks + 8*hsel and k0+16; region boundaries 128/192 are
    //  multiples of 8, so each chunk stays within one source region).
    const int k0 = ks * 32 + hsel * 8;
    const int k1 = k0 + 16;
    const bf16_t* p0 = (k0 < HD) ? (hs + k0)
                     : (k0 < HD + EHD) ? (ee + (k0 - HD)) : (hd + (k0 - HD - EHD));
    const bf16_t* p1 = (k1 < HD) ? (hs + k1)
                     : (k1 < HD + EHD) ? (ee + (k1 - HD)) : (hd + (k1 - HD - EHD));
    U8 au = a_frag_from2(*(const uint4*)p0, *(const uint4*)p1);
    v16bf a = __builtin_bit_cast(v16bf, au);

    const unsigned char* sb = sW + cur * 8192;
#pragma unroll
    for (int nt = 0; nt < 8; ++nt) {
      const unsigned char* bp = sb + ((nt * 16 + m) * 32 + b_k(hsel)) * 2;
      U8 bu = a_frag_from2(*(const uint4*)bp, *(const uint4*)(bp + 32));
      acc[nt] = wmma_bf16(a, __builtin_bit_cast(v16bf, bu), acc[nt]);
    }
    if (ks + 1 < KMSG / 32) {
      wait_async0();     // my next-chunk async copies have landed
      __syncthreads();   // everyone done reading cur + next chunk visible
    }
  }

  // scatter: C/D element (lane, v) = row M = v + 8*hsel, col N = m.
  long long drow[8];
#pragma unroll
  for (int v = 0; v < 8; ++v) drow[v] = ei[NEDGES + tileBase + v + 8 * hsel];
#pragma unroll
  for (int nt = 0; nt < 8; ++nt) {
    int col = nt * 16 + m;
    float b = bias[col];
#pragma unroll
    for (int v = 0; v < 8; ++v)
      atomicAdd(agg + drow[v] * HD + col, acc[nt][v] + b);
  }
}

// ---------------------------------------------------------------------------
// GRU cell, LDS-free: the lane owning a C element does its own sigmoid/tanh.
// Per 16-col block keep gi[r,z,n] and gh[r,z,n] accumulators in VGPRs.
// Also zeroes agg for the next layer.
// ---------------------------------------------------------------------------
__global__ __launch_bounds__(256)
void gru_kernel(const float* __restrict__ agg,     // [N,128] f32
                const float* __restrict__ h_in,    // [N,128] f32 (old h)
                const bf16_t* __restrict__ hb_in,  // [N,128] bf16 (old h)
                const bf16_t* __restrict__ Wi_t,   // [384][128]
                const bf16_t* __restrict__ Wh_t,   // [384][128]
                const float* __restrict__ bi,      // [384]
                const float* __restrict__ bh,      // [384]
                float* __restrict__ h_out,
                bf16_t* __restrict__ hb_out,
                float* __restrict__ agg_zero) {
  const int tid = threadIdx.x, wave = tid >> 5, lane = tid & 31;
  const int m = lane & 15, hsel = lane >> 4;
  const int tile = blockIdx.x * 8 + wave;
  if (tile >= NNODES / 16) return;
  const long long rowm = (long long)tile * 16 + m;

  // Preload A fragments for the full K=128 (4 ks steps), both operands.
  v16bf afr[4], hfr[4];
  const float* ar = agg + rowm * HD;
  const bf16_t* hr = hb_in + rowm * HD;
#pragma unroll
  for (int ks = 0; ks < 4; ++ks) {
    const int k0 = ks * 32 + hsel * 8;
    float4 f0 = *(const float4*)(ar + k0);
    float4 f1 = *(const float4*)(ar + k0 + 4);
    float4 f2 = *(const float4*)(ar + k0 + 16);
    float4 f3 = *(const float4*)(ar + k0 + 20);
    U8 au;
    au.x[0] = pack2(f0.x, f0.y); au.x[1] = pack2(f0.z, f0.w);
    au.x[2] = pack2(f1.x, f1.y); au.x[3] = pack2(f1.z, f1.w);
    au.x[4] = pack2(f2.x, f2.y); au.x[5] = pack2(f2.z, f2.w);
    au.x[6] = pack2(f3.x, f3.y); au.x[7] = pack2(f3.z, f3.w);
    afr[ks] = __builtin_bit_cast(v16bf, au);
    U8 hu = a_frag_from2(*(const uint4*)(hr + k0), *(const uint4*)(hr + k0 + 16));
    hfr[ks] = __builtin_bit_cast(v16bf, hu);
  }

#pragma unroll 1
  for (int nt = 0; nt < 8; ++nt) {
    v8f gi[3], gh[3];
#pragma unroll
    for (int g = 0; g < 3; ++g) { gi[g] = v8f{}; gh[g] = v8f{}; }
#pragma unroll
    for (int g = 0; g < 3; ++g) {
      const int bcol = g * HD + nt * 16 + m;
      const bf16_t* wi = Wi_t + (long long)bcol * HD;
      const bf16_t* wh = Wh_t + (long long)bcol * HD;
#pragma unroll
      for (int ks = 0; ks < 4; ++ks) {
        const int k0 = ks * 32 + b_k(hsel);
        U8 bu = a_frag_from2(*(const uint4*)(wi + k0), *(const uint4*)(wi + k0 + 16));
        U8 cu = a_frag_from2(*(const uint4*)(wh + k0), *(const uint4*)(wh + k0 + 16));
        gi[g] = wmma_bf16(afr[ks], __builtin_bit_cast(v16bf, bu), gi[g]);
        gh[g] = wmma_bf16(hfr[ks], __builtin_bit_cast(v16bf, cu), gh[g]);
      }
    }
    const int col = nt * 16 + m;
    const float bir = bi[col],          bhr = bh[col];
    const float biz = bi[HD + col],     bhz = bh[HD + col];
    const float bin = bi[2 * HD + col], bhn = bh[2 * HD + col];
#pragma unroll
    for (int v = 0; v < 8; ++v) {
      long long r = (long long)tile * 16 + v + 8 * hsel;
      float rr = 1.0f / (1.0f + __expf(-(gi[0][v] + bir + gh[0][v] + bhr)));
      float zz = 1.0f / (1.0f + __expf(-(gi[1][v] + biz + gh[1][v] + bhz)));
      float nn = tanhf(gi[2][v] + bin + rr * (gh[2][v] + bhn));
      float hold = h_in[r * HD + col];
      float hnew = (1.0f - zz) * nn + zz * hold;
      h_out[r * HD + col] = hnew;
      hb_out[r * HD + col] = f2bf(hnew);
      agg_zero[r * HD + col] = 0.0f;
    }
  }
}

// ---------------------------------------------------------------------------
extern "C" void kernel_launch(void* const* d_in, const int* in_sizes, int n_in,
                              void* d_out, int out_size, void* d_ws, size_t ws_size,
                              hipStream_t stream) {
  const float*     x      = (const float*)d_in[0];
  const long long* ei     = (const long long*)d_in[1];   // int64 per reference
  const float*     ea     = (const float*)d_in[2];
  const float*     node_W = (const float*)d_in[3];
  const float*     edge_W = (const float*)d_in[4];
  const float*     msg_W  = (const float*)d_in[5];
  const float*     msg_b  = (const float*)d_in[6];
  const float*     Wi     = (const float*)d_in[7];
  const float*     Wh     = (const float*)d_in[8];
  const float*     bi     = (const float*)d_in[9];
  const float*     bh     = (const float*)d_in[10];
  (void)in_sizes; (void)n_in; (void)out_size; (void)ws_size;

  char* ws = (char*)d_ws;
  size_t off = 0;
  auto alloc = [&](size_t bytes) -> void* {
    void* p = ws + off;
    off += (bytes + 255) & ~(size_t)255;
    return p;
  };
  float*  h_f32  = (float*)alloc((size_t)NNODES * HD * 4);
  float*  agg    = (float*)alloc((size_t)NNODES * HD * 4);
  bf16_t* h_bf   = (bf16_t*)alloc((size_t)NNODES * HD * 2);
  bf16_t* e_bf   = (bf16_t*)alloc((size_t)NEDGES * EHD * 2);
  bf16_t* nodeWt = (bf16_t*)alloc((size_t)HD * FNODE * 2);          // [128][64]
  bf16_t* edgeWt = (bf16_t*)alloc((size_t)EHD * 32 * 2);            // [64][32]
  bf16_t* msgWt  = (bf16_t*)alloc((size_t)NLAYER * HD * KMSG * 2);  // [3][128][320]
  bf16_t* WiT    = (bf16_t*)alloc((size_t)NLAYER * 3 * HD * HD * 2);// [3][384][128]
  bf16_t* WhT    = (bf16_t*)alloc((size_t)NLAYER * 3 * HD * HD * 2);

  // ---- weight prep (transpose + bf16) ----
  auto cvt = [&](const float* s, bf16_t* dst, int K, int Nc, int Kp) {
    int total = Nc * Kp;
    cvt_transpose_kernel<<<(total + 255) / 256, 256, 0, stream>>>(s, dst, K, Nc, Kp);
  };
  cvt(node_W, nodeWt, FNODE, HD, FNODE);        // [64,128] -> [128][64]
  cvt(edge_W, edgeWt, FEDGE, EHD, 32);          // [16,64]  -> [64][32] padded
  for (int l = 0; l < NLAYER; ++l) {
    cvt(msg_W + (size_t)l * KMSG * HD, msgWt + (size_t)l * HD * KMSG, KMSG, HD, KMSG);
    cvt(Wi + (size_t)l * HD * 3 * HD, WiT + (size_t)l * 3 * HD * HD, HD, 3 * HD, HD);
    cvt(Wh + (size_t)l * HD * 3 * HD, WhT + (size_t)l * 3 * HD * HD, HD, 3 * HD, HD);
  }

  // ---- embeddings ----
  embed_edges_kernel<<<NEDGES / 128, 256, 0, stream>>>(ea, edgeWt, e_bf);
  {
    int tiles = NNODES / 16;                     // 3125
    embed_nodes_kernel<<<(tiles + 7) / 8, 256, 0, stream>>>(x, nodeWt, h_f32, h_bf, agg);
  }

  // ---- layers ----
  for (int l = 0; l < NLAYER; ++l) {
    msg_kernel<<<NEDGES / 128, 256, 0, stream>>>(
        h_bf, e_bf, msgWt + (size_t)l * HD * KMSG, msg_b + (size_t)l * HD, ei, agg);
    float* hout = (l == NLAYER - 1) ? (float*)d_out : h_f32;
    int tiles = NNODES / 16;
    gru_kernel<<<(tiles + 7) / 8, 256, 0, stream>>>(
        agg, h_f32, h_bf,
        WiT + (size_t)l * 3 * HD * HD, WhT + (size_t)l * 3 * HD * HD,
        bi + (size_t)l * 3 * HD, bh + (size_t)l * 3 * HD,
        hout, h_bf, agg);
  }
}
